// CausalAttn_35605278884149
// MI455X (gfx1250) — matrix operand order repaired and checked by reading, not compiled, not run on hardware
//
#include <hip/hip_runtime.h>
#include <cstdint>
#include <cstddef>

// Problem constants (match reference)
#define B_   2
#define T_   2048
#define DIM_ 1024
#define H_   16
#define DH_  64
#define BT_  (B_ * T_)

typedef __attribute__((ext_vector_type(16))) __bf16 v16bf;
typedef __attribute__((ext_vector_type(8)))  float  v8f;
typedef __attribute__((ext_vector_type(4)))  unsigned u32x4;
typedef __attribute__((ext_vector_type(2)))  unsigned u32x2;
typedef __attribute__((ext_vector_type(4)))  int      v4i_;

union BF16Frag {
    v16bf          v;
    unsigned short h[16];
    u32x4          q[2];
};

// ---- bf16 conversion helpers (prefer HW packed cvt) -----------------------
#if __has_builtin(__builtin_amdgcn_cvt_pk_bf16_f32)
typedef __attribute__((ext_vector_type(2))) __bf16 v2bf;
__device__ __forceinline__ unsigned pk2bf(float lo, float hi) {
    union { v2bf v; unsigned u; } c;
    c.v = __builtin_amdgcn_cvt_pk_bf16_f32(lo, hi);
    return c.u;
}
__device__ __forceinline__ unsigned short bfbits(float f) {
    union { v2bf v; unsigned short h[2]; } c;
    c.v = __builtin_amdgcn_cvt_pk_bf16_f32(f, 0.0f);
    return c.h[0];
}
#else
__device__ __forceinline__ unsigned short bfbits(float f) {
    union { __bf16 b; unsigned short u; } c;
    c.b = (__bf16)f;
    return c.u;
}
__device__ __forceinline__ unsigned pk2bf(float lo, float hi) {
    return (unsigned)bfbits(lo) | ((unsigned)bfbits(hi) << 16);
}
#endif

// ---- async global -> LDS (CDNA5), guarded ---------------------------------
#if __has_builtin(__builtin_amdgcn_global_load_async_to_lds_b128) && \
    __has_builtin(__builtin_amdgcn_s_wait_asynccnt)
#define HAVE_ASYNC_LDS 1
#else
#define HAVE_ASYNC_LDS 0
#endif

__device__ __forceinline__ v8f vzero8() {
    v8f z;
#pragma unroll
    for (int i = 0; i < 8; ++i) z[i] = 0.0f;
    return z;
}

// ---- 16-lane butterfly reductions via DPP16 (no LDS traffic) --------------
template <int CTRL>
__device__ __forceinline__ float dppf(float x) {
    return __int_as_float(__builtin_amdgcn_update_dpp(
        0, __float_as_int(x), CTRL, 0xf, 0xf, true));
}
__device__ __forceinline__ float rowmax16(float x) {
    x = fmaxf(x, dppf<0xB1>(x));    // quad_perm [1,0,3,2]  : xor 1
    x = fmaxf(x, dppf<0x4E>(x));    // quad_perm [2,3,0,1]  : xor 2
    x = fmaxf(x, dppf<0x141>(x));   // row_half_mirror      : xor 7
    x = fmaxf(x, dppf<0x140>(x));   // row_mirror           : xor 15
    return x;
}
__device__ __forceinline__ float rowsum16(float x) {
    x += dppf<0xB1>(x);
    x += dppf<0x4E>(x);
    x += dppf<0x141>(x);
    x += dppf<0x140>(x);
    return x;
}

// ---------------------------------------------------------------------------
// GEMM: C[M x N] = A[M x K] * W[K x N] + bias  (f32 in, bf16 WMMA)
// MODE 0: f32 output (Cf).
// MODE 1: bf16 output (Ch) with RoPE applied to q/k regions and DH^-0.5
//         folded into q (QKV projection fused epilogue).
// ---------------------------------------------------------------------------
template <int MODE>
__global__ __launch_bounds__(256) void gemm_bf16wmma(
    const float* __restrict__ A, const float* __restrict__ W,
    const float* __restrict__ bias, float* __restrict__ Cf,
    unsigned short* __restrict__ Ch, int M, int N, int K)
{
    constexpr int BM = 64, BN = 128, BK = 32, AP = BK + 8, WP = BK + 8;
    __shared__ __align__(16) unsigned short sAb[BM][AP];   // [row][k]  bf16
    __shared__ __align__(16) unsigned short sWt[BN][WP];   // [col][k]  bf16

    const int tid  = threadIdx.x;
    const int wave = tid >> 5;
    const int lane = tid & 31;
    const int ln   = lane & 15;
    const int hi   = (lane >> 4) & 1;

    const int wm = wave >> 1;   // 0..3
    const int wn = wave & 1;    // 0..1

    const int rowBlock = blockIdx.y * BM;
    const int colBlock = blockIdx.x * BN;

    v8f acc[4];
#pragma unroll
    for (int t = 0; t < 4; ++t) acc[t] = vzero8();

    const int kA0 = hi ? 8 : 0;
    const int kB0 = hi ? 16 : 0;

    for (int k0 = 0; k0 < K; k0 += BK) {
        // ---- A tile 64x32: f32 -> bf16 packed, row-major ----
#pragma unroll
        for (int q = 0; q < 2; ++q) {
            int i4 = tid + q * 256;                 // 0..511 float4 chunks
            int r = i4 >> 3, c4 = (i4 & 7) << 2;
            float4 g = *reinterpret_cast<const float4*>(
                &A[(size_t)(rowBlock + r) * K + k0 + c4]);
            u32x2 p;
            p[0] = pk2bf(g.x, g.y);
            p[1] = pk2bf(g.z, g.w);
            *reinterpret_cast<u32x2*>(&sAb[r][c4]) = p;
        }
        // ---- W tile 32x128: f32 -> bf16 packed, TRANSPOSED to [col][k] ----
#pragma unroll
        for (int q = 0; q < 2; ++q) {
            int mt = tid + q * 256;                 // 0..511
            int kp = mt & 15, cg = mt >> 4;         // k = 2*kp, cols cg*4..+3
            const float4 g0 = *reinterpret_cast<const float4*>(
                &W[(size_t)(k0 + 2 * kp) * N + colBlock + cg * 4]);
            const float4 g1 = *reinterpret_cast<const float4*>(
                &W[(size_t)(k0 + 2 * kp + 1) * N + colBlock + cg * 4]);
            *reinterpret_cast<unsigned*>(&sWt[cg * 4 + 0][2 * kp]) = pk2bf(g0.x, g1.x);
            *reinterpret_cast<unsigned*>(&sWt[cg * 4 + 1][2 * kp]) = pk2bf(g0.y, g1.y);
            *reinterpret_cast<unsigned*>(&sWt[cg * 4 + 2][2 * kp]) = pk2bf(g0.z, g1.z);
            *reinterpret_cast<unsigned*>(&sWt[cg * 4 + 3][2 * kp]) = pk2bf(g0.w, g1.w);
        }
        if (k0 + BK < K) {  // CDNA5 global_prefetch_b8 for next K tile
            __builtin_prefetch(&A[(size_t)(rowBlock + (tid >> 2)) * K + k0 + BK], 0, 1);
            __builtin_prefetch(&W[(size_t)(k0 + BK + (tid >> 5)) * N + colBlock + (tid & 31) * 4], 0, 1);
        }
        __syncthreads();

        BF16Frag af;
        af.q[0] = *reinterpret_cast<const u32x4*>(&sAb[wm * 16 + ln][kA0]);
        af.q[1] = *reinterpret_cast<const u32x4*>(&sAb[wm * 16 + ln][kA0 + 16]);
#pragma unroll
        for (int t = 0; t < 4; ++t) {
            BF16Frag bf;
            const unsigned short* bp = &sWt[wn * 64 + t * 16 + ln][kA0];
            bf.q[0] = *reinterpret_cast<const u32x4*>(bp);
            bf.q[1] = *reinterpret_cast<const u32x4*>(bp + 16);
            acc[t] = __builtin_amdgcn_wmma_f32_16x16x32_bf16(
                false, af.v, false, bf.v, (short)0, acc[t], false, false);
        }
        __syncthreads();
    }

    if (MODE == 0) {
#pragma unroll
        for (int t = 0; t < 4; ++t)
#pragma unroll
            for (int r = 0; r < 8; ++r) {
                int row = rowBlock + wm * 16 + r + hi * 8;
                int col = colBlock + wn * 64 + t * 16 + ln;
                Cf[(size_t)row * N + col] = acc[t][r] + bias[col];
            }
    } else {
        // QKV epilogue: bf16 out, RoPE on q/k, q pre-scaled by DH^-0.5.
        const int colBase = colBlock + wn * 64;   // 64-col strip == one head
        const int region  = colBase >> 10;        // 0=q, 1=k, 2=v
        if (region <= 1) {
            const float qscale = (region == 0) ? 0.125f : 1.0f;
#pragma unroll
            for (int t2 = 0; t2 < 2; ++t2) {
                const int d = t2 * 16 + ln;        // 0..31
                const float inv = __powf(10000.0f, -(float)d * (1.0f / 32.0f));
#pragma unroll
                for (int r = 0; r < 8; ++r) {
                    const int row = rowBlock + wm * 16 + r + hi * 8;
                    const int pos = row & (T_ - 1);
                    float s, c;
                    __sincosf((float)pos * inv, &s, &c);
                    const float a  = acc[t2][r]     + bias[colBase + d];
                    const float b2 = acc[t2 + 2][r] + bias[colBase + d + 32];
                    const float ra = (a * c - b2 * s) * qscale;
                    const float rb = (a * s + b2 * c) * qscale;
                    Ch[(size_t)row * N + colBase + d]      = bfbits(ra);
                    Ch[(size_t)row * N + colBase + d + 32] = bfbits(rb);
                }
            }
        } else {  // v: plain convert
#pragma unroll
            for (int t = 0; t < 4; ++t)
#pragma unroll
                for (int r = 0; r < 8; ++r) {
                    int row = rowBlock + wm * 16 + r + hi * 8;
                    int col = colBase + t * 16 + ln;
                    Ch[(size_t)row * N + col] = bfbits(acc[t][r] + bias[col]);
                }
        }
    }
}

// ---------------------------------------------------------------------------
// Flash attention (causal, online softmax) over bf16 qkv.
// Block = (b, h, 128 query rows); 8 waves each own 16 rows end-to-end.
// K tile row-major bf16 (async-copied to LDS when available);
// V tile transposed bf16; P via wave-private slab. Row stats use DPP16.
// ---------------------------------------------------------------------------
__global__ __launch_bounds__(256) void flash_attn_bf16wmma(
    const unsigned short* __restrict__ qkvh, float* __restrict__ Y)
{
    constexpr int KP = 72;  // u16 pitch (144 B, 16B-aligned vectors)
    __shared__ __align__(16) unsigned short sKb[64][KP];      // [key][dim]
    __shared__ __align__(16) unsigned short sVt[64][KP];      // [dim][key]
    __shared__ __align__(16) unsigned short sP[8][16][KP];    // per-wave P

    const int tid  = threadIdx.x;
    const int wave = tid >> 5;
    const int lane = tid & 31;
    const int ln   = lane & 15;
    const int hi   = (lane >> 4) & 1;

    const int qBase = blockIdx.x * 128;
    const int h = blockIdx.y;
    const int b = blockIdx.z;
    const int wr = qBase + wave * 16;

    const int kA0 = hi ? 8 : 0;
    const int kB0 = hi ? 16 : 0;
    const size_t rowPitch = 3 * DIM_;

    // Q fragments: direct bf16 vector loads (scale already folded in)
    BF16Frag qa[2];
    {
        const unsigned short* qp =
            qkvh + (size_t)(b * T_ + wr + ln) * rowPitch + h * DH_;
#pragma unroll
        for (int s0 = 0; s0 < 2; ++s0) {
            qa[s0].q[0] = *reinterpret_cast<const u32x4*>(qp + s0 * 32 + kA0);
            qa[s0].q[1] = *reinterpret_cast<const u32x4*>(qp + s0 * 32 + kA0 + 16);
        }
    }

    v8f o[4];
#pragma unroll
    for (int t = 0; t < 4; ++t) o[t] = vzero8();
    float mrow[8], lrow[8];    // lrow is LANE-PARTIAL; reduced once at the end
#pragma unroll
    for (int r = 0; r < 8; ++r) { mrow[r] = -__builtin_inff(); lrow[r] = 0.0f; }

    const int nJ = (qBase >> 6) + 2;
    for (int j = 0; j < nJ; ++j) {
        __syncthreads();   // previous iteration's K/V reads complete
        // ---- K tile: 64x64 bf16 row-major copy (async to LDS if possible) --
#pragma unroll
        for (int q = 0; q < 2; ++q) {
            int c = tid + q * 256;              // 0..511 chunks of 8 bf16
            int key = c >> 3, g8 = (c & 7) << 3;
            const unsigned short* src =
                qkvh + (size_t)(b * T_ + j * 64 + key) * rowPitch + DIM_ + h * DH_ + g8;
#if HAVE_ASYNC_LDS
            __builtin_amdgcn_global_load_async_to_lds_b128(
                (__attribute__((address_space(1))) v4i_*)src,
                (__attribute__((address_space(3))) v4i_*)&sKb[key][g8], 0, 0);
#else
            *reinterpret_cast<u32x4*>(&sKb[key][g8]) =
                *reinterpret_cast<const u32x4*>(src);
#endif
        }
        // ---- V tile transposed: key-pair x 8-dim micro-tile per thread ----
        {
            int kp = tid >> 3, dg = (tid & 7) << 3;   // keys 2kp,2kp+1; dims dg..+7
            const unsigned short* v0 =
                qkvh + (size_t)(b * T_ + j * 64 + 2 * kp) * rowPitch + 2 * DIM_ + h * DH_ + dg;
            union { u32x4 v; unsigned short h[8]; } va, vb2;
            va.v  = *reinterpret_cast<const u32x4*>(v0);
            vb2.v = *reinterpret_cast<const u32x4*>(v0 + rowPitch);
#pragma unroll
            for (int i = 0; i < 8; ++i) {
                unsigned pk = (unsigned)va.h[i] | ((unsigned)vb2.h[i] << 16);
                *reinterpret_cast<unsigned*>(&sVt[dg + i][2 * kp]) = pk;
            }
        }
#if HAVE_ASYNC_LDS
        __builtin_amdgcn_s_wait_asynccnt(0);
#endif
        __syncthreads();

        const bool active   = (j * 64) <= (wr + 15);        // wave-uniform
        const bool needMask = (j * 64 + 63) > wr;           // wave-uniform
        if (active) {
            // ---- S = Q K^T : 16 rows x 64 keys ----
            v8f st[4];
#pragma unroll
            for (int t = 0; t < 4; ++t) {
                v8f s = vzero8();
#pragma unroll
                for (int s0 = 0; s0 < 2; ++s0) {
                    BF16Frag kb;
                    const unsigned short* kpp = &sKb[t * 16 + ln][s0 * 32 + kA0];
                    kb.q[0] = *reinterpret_cast<const u32x4*>(kpp);
                    kb.q[1] = *reinterpret_cast<const u32x4*>(kpp + 16);
                    s = __builtin_amdgcn_wmma_f32_16x16x32_bf16(
                        false, qa[s0].v, false, kb.v, (short)0, s, false, false);
                }
                st[t] = s;
            }

            // ---- causal mask + online softmax (DPP16 row max) ----
#pragma unroll
            for (int r = 0; r < 8; ++r) {
                const int qi = wr + r + hi * 8;
                float mx = -__builtin_inff();
                if (needMask) {
#pragma unroll
                    for (int t = 0; t < 4; ++t) {
                        const int kj = j * 64 + t * 16 + ln;
                        float sv = (kj <= qi) ? st[t][r] : -__builtin_inff();
                        st[t][r] = sv;
                        mx = fmaxf(mx, sv);
                    }
                } else {
#pragma unroll
                    for (int t = 0; t < 4; ++t) mx = fmaxf(mx, st[t][r]);
                }
                mx = rowmax16(mx);                      // lane-uniform row max
                const float mnew = fmaxf(mrow[r], mx);
                const float corr = __expf(mrow[r] - mnew);
                float ls = 0.0f;
#pragma unroll
                for (int t = 0; t < 4; ++t) {
                    float pv = __expf(st[t][r] - mnew);
                    st[t][r] = pv;
                    ls += pv;
                }
                lrow[r] = lrow[r] * corr + ls;          // lane-partial sum
                mrow[r] = mnew;
#pragma unroll
                for (int t = 0; t < 4; ++t) o[t][r] *= corr;
            }

            // ---- P -> wave-private LDS slab (C-layout -> A-layout) ----
#pragma unroll
            for (int t = 0; t < 4; ++t)
#pragma unroll
                for (int r = 0; r < 8; ++r)
                    sP[wave][r + hi * 8][t * 16 + ln] = bfbits(st[t][r]);
            // same-wave DS ops are in-order: no barrier needed before re-read

            BF16Frag pa[2];
#pragma unroll
            for (int s0 = 0; s0 < 2; ++s0) {
                const unsigned short* pp = &sP[wave][ln][s0 * 32 + kA0];
                pa[s0].q[0] = *reinterpret_cast<const u32x4*>(pp);
                pa[s0].q[1] = *reinterpret_cast<const u32x4*>(pp + 16);
            }
            // ---- O += P V ----
#pragma unroll
            for (int t = 0; t < 4; ++t) {
#pragma unroll
                for (int s0 = 0; s0 < 2; ++s0) {
                    BF16Frag vb;
                    const unsigned short* vp = &sVt[t * 16 + ln][s0 * 32 + kA0];
                    vb.q[0] = *reinterpret_cast<const u32x4*>(vp);
                    vb.q[1] = *reinterpret_cast<const u32x4*>(vp + 16);
                    o[t] = __builtin_amdgcn_wmma_f32_16x16x32_bf16(
                        false, pa[s0].v, false, vb.v, (short)0, o[t], false, false);
                }
            }
        }
    }

    // normalize (single final 16-lane sum per row) and store y as f32
#pragma unroll
    for (int r = 0; r < 8; ++r) {
        const int qi = wr + r + hi * 8;
        const float nl = 1.0f / rowsum16(lrow[r]);
#pragma unroll
        for (int t = 0; t < 4; ++t)
            Y[(size_t)(b * T_ + qi) * DIM_ + h * DH_ + t * 16 + ln] = o[t][r] * nl;
    }
}

// ---------------------------------------------------------------------------
// Host-side launcher
// ---------------------------------------------------------------------------
extern "C" void kernel_launch(void* const* d_in, const int* in_sizes, int n_in,
                              void* d_out, int out_size, void* d_ws, size_t ws_size,
                              hipStream_t stream)
{
    const float* x     = (const float*)d_in[0];
    // d_in[1]: mask — unused, causal mask computed analytically
    const float* W_qkv = (const float*)d_in[2];
    const float* b_qkv = (const float*)d_in[3];
    const float* W_out = (const float*)d_in[4];
    const float* b_out = (const float*)d_in[5];
    float* out = (float*)d_out;

    unsigned short* qkvh = (unsigned short*)d_ws;              // 24 MB bf16
    float* y = (float*)(qkvh + (size_t)BT_ * 3 * DIM_);        // 16 MB f32

    // 1) QKV projection + bias + RoPE + bf16 pack (fused epilogue)
    gemm_bf16wmma<1><<<dim3(3 * DIM_ / 128, BT_ / 64), 256, 0, stream>>>(
        x, W_qkv, b_qkv, nullptr, qkvh, BT_, 3 * DIM_, DIM_);
    // 2) causal flash attention on bf16 qkv
    flash_attn_bf16wmma<<<dim3(T_ / 128, H_, B_), 256, 0, stream>>>(qkvh, y);
    // 3) output projection (f32 out)
    gemm_bf16wmma<0><<<dim3(DIM_ / 128, BT_ / 64), 256, 0, stream>>>(
        y, W_out, b_out, out, nullptr, BT_, DIM_, DIM_);
}
